// GINEModel_84207128805738
// MI455X (gfx1250) — matrix operand order repaired and checked by reading, not compiled, not run on hardware
//
#include <hip/hip_runtime.h>
#include <hip/hip_bf16.h>

// ---------------------------------------------------------------------------
// Types for gfx1250 WMMA (wave32): 16x16x32 bf16 -> f32
// ---------------------------------------------------------------------------
typedef __bf16 bf16_t;
typedef __attribute__((ext_vector_type(16))) __bf16 v16bf;
typedef __attribute__((ext_vector_type(4)))  __bf16 v4bf;
typedef __attribute__((ext_vector_type(8)))  float  v8f;

static __device__ __forceinline__ bf16_t f2bf(float f) {
    // round-to-nearest-even f32 -> bf16
    unsigned u = __builtin_bit_cast(unsigned, f);
    u += 0x7FFFu + ((u >> 16) & 1u);
    unsigned short s = (unsigned short)(u >> 16);
    return __builtin_bit_cast(bf16_t, s);
}

static __device__ __forceinline__ void atomic_add_f32(float* p, float v) {
    __hip_atomic_fetch_add(p, v, __ATOMIC_RELAXED, __HIP_MEMORY_SCOPE_AGENT);
}

// ---------------------------------------------------------------------------
// Utility kernels
// ---------------------------------------------------------------------------
__global__ void fill_zero_kernel(float* __restrict__ p, int n) {
    int i = blockIdx.x * blockDim.x + threadIdx.x;
    if (i < n) p[i] = 0.0f;
}

// ---------------------------------------------------------------------------
// Weight packing: W [K, Ncols] f32 (row-major) -> fragment-ordered bf16.
// Layout: fragment tile (kt, jt) occupies 512 bf16:
//   Bpack[(kt*NJ + jt)*512 + L*16 + e]  holds  W[kt*32 + (L>>4)*16 + e][jt*16 + (L&15)]
// so lane L of a wave loads its whole 16x(bf16) B fragment from one
// contiguous 32-byte span (two global_load_b128).
// ---------------------------------------------------------------------------
__global__ void pack_b_kernel(const float* __restrict__ W, bf16_t* __restrict__ Bpack,
                              int K, int Ncols) {
    int idx = blockIdx.x * blockDim.x + threadIdx.x;
    if (idx >= K * Ncols) return;
    const int NJ = Ncols >> 4;
    int frag   = idx >> 9;        // which (kt, jt) tile
    int within = idx & 511;
    int L = within >> 4;
    int e = within & 15;
    int kt = frag / NJ;
    int jt = frag - kt * NJ;
    int k   = kt * 32 + (L >> 4) * 16 + e;
    int col = jt * 16 + (L & 15);
    Bpack[idx] = f2bf(W[(size_t)k * Ncols + col]);
}

// ---------------------------------------------------------------------------
// Edge kernel: one wave32 per edge.
//   m = relu(x[src] + edge_attr @ We + be);  agg[dst] += m  (atomic f32)
// H = feature width (64 or 256), We is [8, H].
// ---------------------------------------------------------------------------
template <int H>
__global__ __launch_bounds__(256)
void edge_conv_kernel(const float* __restrict__ xin, const int* __restrict__ ei,
                      const float* __restrict__ ea, const float* __restrict__ We,
                      const float* __restrict__ be, float* __restrict__ agg, int E) {
    constexpr int CPL = H / 32;  // columns per lane
    int wid  = blockIdx.x * 8 + (threadIdx.x >> 5);
    if (wid >= E) return;
    int lane = threadIdx.x & 31;
    int s = ei[wid];
    int d = ei[E + wid];
    // hint the gathered row into cache while we do the 8-wide edge linear
    __builtin_prefetch(&xin[(size_t)s * H], 0, 3);
    const float4* ea4 = reinterpret_cast<const float4*>(ea);
    float4 e0 = ea4[(size_t)wid * 2 + 0];
    float4 e1 = ea4[(size_t)wid * 2 + 1];
    float a[8] = {e0.x, e0.y, e0.z, e0.w, e1.x, e1.y, e1.z, e1.w};
#pragma unroll
    for (int j = 0; j < CPL; ++j) {
        int c = j * 32 + lane;
        float v = be[c];
#pragma unroll
        for (int k = 0; k < 8; ++k) v = fmaf(a[k], We[k * H + c], v);
        v += xin[(size_t)s * H + c];
        v = fmaxf(v, 0.0f);
        atomic_add_f32(&agg[(size_t)d * H + c], v);
    }
}

// ---------------------------------------------------------------------------
// bf16 WMMA GEMM:  C[M,Ncols] = act(A) @ B + bias
//   act(A) = (epsp ? (1+*epsp) : 1) * A  (+ A2 if A2 != nullptr)
// Block: 256 threads = 8 waves. Block tile = 128(M) x (16*NT)(N).
// A is staged per 32-wide K step into LDS *already in fragment order* (bf16),
// so each lane pulls its A fragment with one contiguous 32B LDS read.
// B comes pre-packed in fragment order (see pack_b_kernel).
// K is a template parameter so the K-tile loop fully unrolls.
// ---------------------------------------------------------------------------
template <int K, int NT>
__global__ __launch_bounds__(256)
void gemm_bf16_wmma_kernel(const float* __restrict__ A, const float* __restrict__ A2,
                           const float* __restrict__ epsp,
                           const bf16_t* __restrict__ Bpack, const float* __restrict__ bias,
                           float* __restrict__ C, int M, int Ncols) {
    // 8 wave-tiles x 512 bf16, fragment-ordered
    __shared__ alignas(32) bf16_t sA[8 * 512];

    const int tid  = threadIdx.x;
    const int wave = tid >> 5;
    const int lane = tid & 31;
    const int r    = lane & 15;   // col within C tile
    const int half = lane >> 4;
    const int rowBase = blockIdx.x * 128;
    const int jt0     = blockIdx.y * NT;      // first 16-col tile index
    const int NJ      = Ncols >> 4;

    float alpha = 1.0f;
    if (epsp) alpha += epsp[0];

    v8f acc[NT];
#pragma unroll
    for (int j = 0; j < NT; ++j) acc[j] = (v8f){0.f, 0.f, 0.f, 0.f, 0.f, 0.f, 0.f, 0.f};

#pragma unroll
    for (int kt = 0; kt < K / 32; ++kt) {
        // ---- cooperative stage: 128 rows x 32 K, float4 loads, fused update,
        //      bf16-converted and scattered into fragment order (8B ds stores)
        for (int i = tid; i < 128 * 8; i += 256) {
            int row  = i >> 3;
            int k4   = (i & 7) * 4;
            int grow = rowBase + row;
            float4 v = make_float4(0.f, 0.f, 0.f, 0.f);
            if (grow < M) {
                v = *(const float4*)&A[(size_t)grow * K + kt * 32 + k4];
                if (A2) {
                    float4 u = *(const float4*)&A2[(size_t)grow * K + kt * 32 + k4];
                    v.x = alpha * v.x + u.x;
                    v.y = alpha * v.y + u.y;
                    v.z = alpha * v.z + u.z;
                    v.w = alpha * v.w + u.w;
                }
            }
            // fragment-order position: lane L = hh*16 + (row&15), element e0..e0+3
            int w  = row >> 4;
            int rr = row & 15;
            int hh = k4 >> 4;
            int e0 = k4 & 15;
            v4bf pk;
            pk[0] = f2bf(v.x);
            pk[1] = f2bf(v.y);
            pk[2] = f2bf(v.z);
            pk[3] = f2bf(v.w);
            *(v4bf*)&sA[w * 512 + (hh * 16 + rr) * 16 + e0] = pk;
        }
        __syncthreads();

        // ---- one contiguous 32B LDS read per lane = whole A fragment
        v16bf afrag = *(const v16bf*)&sA[wave * 512 + lane * 16];

#pragma unroll
        for (int j = 0; j < NT; ++j) {
            // one contiguous 32B global read per lane = whole B fragment
            v16bf bfrag =
                *(const v16bf*)&Bpack[((size_t)(kt * NJ + jt0 + j)) * 512 + lane * 16];
            acc[j] = __builtin_amdgcn_wmma_f32_16x16x32_bf16(
                false, afrag, false, bfrag, (short)0, acc[j], false, false);
        }
        __syncthreads();
    }

    // ---- epilogue: bias add, f32 store (C/D layout: VGPR v -> M = half*8+v)
#pragma unroll
    for (int j = 0; j < NT; ++j) {
        int c = (jt0 + j) * 16 + r;
        float bv = bias[c];
#pragma unroll
        for (int vr = 0; vr < 8; ++vr) {
            int grow = rowBase + wave * 16 + half * 8 + vr;
            if (grow < M) C[(size_t)grow * Ncols + c] = acc[j][vr] + bv;
        }
    }
}

// ---------------------------------------------------------------------------
// BatchNorm (training-mode batch stats), two-pass.
// ---------------------------------------------------------------------------
__global__ void bn_stats_kernel(const float* __restrict__ T, float* __restrict__ sums,
                                float* __restrict__ sq, int M, int H) {
    int c  = threadIdx.x;          // blockDim.x == H (128 or 256)
    int r0 = blockIdx.x * 128;
    float s = 0.0f, q = 0.0f;
    for (int i = 0; i < 128; ++i) {
        int r = r0 + i;
        if (r < M) {
            float v = T[(size_t)r * H + c];
            s += v;
            q += v * v;
        }
    }
    atomic_add_f32(&sums[c], s);
    atomic_add_f32(&sq[c], q);
}

__global__ void bn_apply_relu_kernel(float* __restrict__ T, const float* __restrict__ sums,
                                     const float* __restrict__ sq,
                                     const float* __restrict__ g, const float* __restrict__ b,
                                     int M, int H) {
    int i = blockIdx.x * blockDim.x + threadIdx.x;
    if (i >= M * H) return;
    int c = i & (H - 1);  // H is a power of two
    float invM = 1.0f / (float)M;
    float mu   = sums[c] * invM;
    float var  = sq[c] * invM - mu * mu;
    float inv  = rsqrtf(var + 1e-5f);
    float v    = g[c] * (T[i] - mu) * inv + b[c];
    T[i] = fmaxf(v, 0.0f);
}

// ---------------------------------------------------------------------------
// Head: out[n] = h2[n,:128] . Wout + bout   (one wave32 per row)
// ---------------------------------------------------------------------------
__global__ __launch_bounds__(256)
void head_kernel(const float* __restrict__ h, const float* __restrict__ Wout,
                 const float* __restrict__ bout, float* __restrict__ out, int M) {
    int wid  = blockIdx.x * 8 + (threadIdx.x >> 5);
    int lane = threadIdx.x & 31;
    if (wid >= M) return;
    float s = 0.0f;
#pragma unroll
    for (int k = lane; k < 128; k += 32) s = fmaf(h[(size_t)wid * 128 + k], Wout[k], s);
#pragma unroll
    for (int off = 16; off > 0; off >>= 1) s += __shfl_down(s, off, 32);
    if (lane == 0) out[wid] = s + bout[0];
}

// ---------------------------------------------------------------------------
// Host launcher
// ---------------------------------------------------------------------------
extern "C" void kernel_launch(void* const* d_in, const int* in_sizes, int n_in,
                              void* d_out, int out_size, void* d_ws, size_t ws_size,
                              hipStream_t stream) {
    (void)n_in; (void)out_size; (void)ws_size;

    const float* x    = (const float*)d_in[0];
    const int*   ei   = (const int*)d_in[1];
    const float* ea   = (const float*)d_in[2];
    const float* eps1 = (const float*)d_in[3];
    const float* We1  = (const float*)d_in[4];
    const float* be1  = (const float*)d_in[5];
    const float* W11  = (const float*)d_in[6];
    const float* b11  = (const float*)d_in[7];
    const float* g11  = (const float*)d_in[8];
    const float* bt11 = (const float*)d_in[9];
    const float* W12  = (const float*)d_in[10];
    const float* b12  = (const float*)d_in[11];
    const float* g1   = (const float*)d_in[12];
    const float* bt1  = (const float*)d_in[13];
    const float* eps2 = (const float*)d_in[14];
    const float* We2  = (const float*)d_in[15];
    const float* be2  = (const float*)d_in[16];
    const float* W21  = (const float*)d_in[17];
    const float* b21  = (const float*)d_in[18];
    const float* g21  = (const float*)d_in[19];
    const float* bt21 = (const float*)d_in[20];
    const float* W22  = (const float*)d_in[21];
    const float* b22  = (const float*)d_in[22];
    const float* g2   = (const float*)d_in[23];
    const float* bt2  = (const float*)d_in[24];
    const float* Wout = (const float*)d_in[25];
    const float* bout = (const float*)d_in[26];

    const int N = in_sizes[0] / 64;  // x: [N, 64]
    const int E = in_sizes[2] / 8;   // edge_attr: [E, 8]
    constexpr int F_IN = 64, H1 = 256, H2 = 128;

    // ---- workspace carve (256B-aligned regions) ----
    size_t off = 0;
    auto carve = [&](size_t bytes) -> void* {
        void* p = (char*)d_ws + off;
        off += (bytes + 255) & ~(size_t)255;
        return p;
    };
    bf16_t* W11b = (bf16_t*)carve((size_t)F_IN * H1 * sizeof(bf16_t));
    bf16_t* W12b = (bf16_t*)carve((size_t)H1 * H1 * sizeof(bf16_t));
    bf16_t* W21b = (bf16_t*)carve((size_t)H1 * H2 * sizeof(bf16_t));
    bf16_t* W22b = (bf16_t*)carve((size_t)H2 * H2 * sizeof(bf16_t));
    float*  ssum = (float*)carve(256 * sizeof(float));
    float*  ssq  = (float*)carve(256 * sizeof(float));
    float*  agg1 = (float*)carve((size_t)N * F_IN * sizeof(float));
    float*  t1   = (float*)carve((size_t)N * H1 * sizeof(float));  // reused as agg2
    float*  t2   = (float*)carve((size_t)N * H1 * sizeof(float));  // h1
    float*  t3   = (float*)carve((size_t)N * H2 * sizeof(float));
    float*  t4   = (float*)carve((size_t)N * H2 * sizeof(float));
    float*  agg2 = t1;
    float*  outp = (float*)d_out;

    auto zero = [&](float* p, int n) {
        fill_zero_kernel<<<(n + 255) / 256, 256, 0, stream>>>(p, n);
    };
    auto pack = [&](const float* s, bf16_t* d, int K, int Nc) {
        pack_b_kernel<<<(K * Nc + 255) / 256, 256, 0, stream>>>(s, d, K, Nc);
    };

    const int gEdge = (E + 7) / 8;
    const int gM    = (N + 127) / 128;

    // ---- pack weights into fragment-ordered bf16 ----
    pack(W11, W11b, F_IN, H1);
    pack(W12, W12b, H1, H1);
    pack(W21, W21b, H1, H2);
    pack(W22, W22b, H2, H2);

    // ================= conv1 =================
    zero(agg1, N * F_IN);
    edge_conv_kernel<F_IN><<<gEdge, 256, 0, stream>>>(x, ei, ea, We1, be1, agg1, E);

    // t1 = ((1+eps1)*x + agg1) @ W11 + b11      [N,256]
    gemm_bf16_wmma_kernel<F_IN, 4><<<dim3(gM, H1 / 64), 256, 0, stream>>>(
        x, agg1, eps1, W11b, b11, t1, N, H1);
    zero(ssum, H1); zero(ssq, H1);
    bn_stats_kernel<<<gM, H1, 0, stream>>>(t1, ssum, ssq, N, H1);
    bn_apply_relu_kernel<<<((size_t)N * H1 + 255) / 256, 256, 0, stream>>>(
        t1, ssum, ssq, g11, bt11, N, H1);

    // t2 = t1 @ W12 + b12                       [N,256]
    gemm_bf16_wmma_kernel<H1, 4><<<dim3(gM, H1 / 64), 256, 0, stream>>>(
        t1, nullptr, nullptr, W12b, b12, t2, N, H1);
    zero(ssum, H1); zero(ssq, H1);
    bn_stats_kernel<<<gM, H1, 0, stream>>>(t2, ssum, ssq, N, H1);
    bn_apply_relu_kernel<<<((size_t)N * H1 + 255) / 256, 256, 0, stream>>>(
        t2, ssum, ssq, g1, bt1, N, H1);  // t2 == h1

    // ================= conv2 =================
    zero(agg2, N * H1);
    edge_conv_kernel<H1><<<gEdge, 256, 0, stream>>>(t2, ei, ea, We2, be2, agg2, E);

    // t3 = ((1+eps2)*h1 + agg2) @ W21 + b21     [N,128]
    gemm_bf16_wmma_kernel<H1, 4><<<dim3(gM, H2 / 64), 256, 0, stream>>>(
        t2, agg2, eps2, W21b, b21, t3, N, H2);
    zero(ssum, H2); zero(ssq, H2);
    bn_stats_kernel<<<gM, H2, 0, stream>>>(t3, ssum, ssq, N, H2);
    bn_apply_relu_kernel<<<((size_t)N * H2 + 255) / 256, 256, 0, stream>>>(
        t3, ssum, ssq, g21, bt21, N, H2);

    // t4 = t3 @ W22 + b22                       [N,128]
    gemm_bf16_wmma_kernel<H2, 4><<<dim3(gM, H2 / 64), 256, 0, stream>>>(
        t3, nullptr, nullptr, W22b, b22, t4, N, H2);
    zero(ssum, H2); zero(ssq, H2);
    bn_stats_kernel<<<gM, H2, 0, stream>>>(t4, ssum, ssq, N, H2);
    bn_apply_relu_kernel<<<((size_t)N * H2 + 255) / 256, 256, 0, stream>>>(
        t4, ssum, ssq, g2, bt2, N, H2);

    // ================= head =================
    head_kernel<<<(N + 7) / 8, 256, 0, stream>>>(t4, Wout, bout, outp, N);
}
